// EvolveGCNIIO_16106127360502
// MI455X (gfx1250) — matrix operand order repaired and checked by reading, not compiled, not run on hardware
//
#include <hip/hip_runtime.h>

#define C_DIM   128
#define FOURC   512
#define N_NODES 50000
#define E_EDGES 600000
#define NTILE   8            // C_DIM / 16 column tiles
#define KT128   4            // 128 / 32 k-tiles

typedef __attribute__((ext_vector_type(16))) __bf16 v16bf;
typedef __attribute__((ext_vector_type(8)))  float  v8f;

__device__ __forceinline__ float sigmoidf_(float x) { return 1.0f / (1.0f + __expf(-x)); }

__device__ __forceinline__ void atomAddF(float* p, float v) {
    unsafeAtomicAdd(p, v);   // global_atomic_add_f32
}

// ---- WMMA fragment helpers (CDNA5 16x16x32 bf16, wave32 layouts) ----------

// A: 16x32 bf16 from fp32 row-major source (lda = C_DIM)
__device__ __forceinline__ v16bf load_a_f32(const float* __restrict__ A, int m0, int k0, int lane) {
    v16bf a;
    const int row = m0 + (lane & 15);
    const int kb  = k0 + ((lane & 16) ? 8 : 0);
    const float* p = A + (size_t)row * C_DIM + kb;
#pragma unroll
    for (int j = 0; j < 8; ++j) a[j]     = (__bf16)p[j];        // K = kb..kb+7
#pragma unroll
    for (int j = 0; j < 8; ++j) a[8 + j] = (__bf16)p[16 + j];   // K = kb+16..kb+23
    return a;
}

// B: fragment-major packed layout; each lane reads 32 contiguous bytes
// packed index = ((ntile*KT + ktile) * 32 + lane) * 16 + j
__device__ __forceinline__ v16bf load_b_frag(const __bf16* __restrict__ P, int ntile, int ktile, int lane) {
    const __bf16* p = P + (((size_t)(ntile * KT128 + ktile) << 5 | lane) << 4);
    union { v16bf v; uint4 q[2]; } u;
    u.q[0] = ((const uint4*)p)[0];
    u.q[1] = ((const uint4*)p)[1];
    return u.v;
}

__device__ __forceinline__ v8f wmma_bf16(v16bf a, v16bf b, v8f c) {
    return __builtin_amdgcn_wmma_f32_16x16x32_bf16(false, a, false, b, (short)0, c, false, false);
}

// Pack fp32 weight (K x N, optionally transposed source) into fragment-major bf16.
// trans=0: B[k][n] = src[k*N + n];  trans=1: B[k][n] = src[n*K + k]  (for Wih^T)
__global__ void pack_b_kernel(const float* __restrict__ src, __bf16* __restrict__ dst,
                              int K, int N, int trans) {
    int i = blockIdx.x * blockDim.x + threadIdx.x;
    if (i >= K * N) return;
    const int frag = i >> 9;
    const int r    = i & 511;
    const int lane = r >> 4;
    const int j    = r & 15;
    const int KT   = K >> 5;
    const int ktile = frag % KT;
    const int ntile = frag / KT;
    const int k = ktile * 32 + ((lane & 16) ? 8 : 0) + (j & 7) + ((j & 8) ? 16 : 0);
    const int n = ntile * 16 + (lane & 15);
    const float v = trans ? src[(size_t)n * K + k] : src[(size_t)k * N + n];
    dst[i] = (__bf16)v;
}

// ---- GEMM kernels ---------------------------------------------------------

// Out[M x C] = X[M x C] @ W   (W pre-packed fragment-major bf16)
__global__ __launch_bounds__(256) void gemm_xw_kernel(const float* __restrict__ X,
                                                      const __bf16* __restrict__ Wb,
                                                      float* __restrict__ Out, int M) {
    const int wave = (blockIdx.x * blockDim.x + threadIdx.x) >> 5;
    const int lane = threadIdx.x & 31;
    const int m0 = (wave / NTILE) * 16;
    const int nt = wave % NTILE;
    if (m0 >= M) return;
    v8f acc = {};
#pragma unroll
    for (int kt = 0; kt < KT128; ++kt) {
        v16bf a = load_a_f32(X, m0, kt * 32, lane);
        v16bf b = load_b_frag(Wb, nt, kt, lane);
        acc = wmma_bf16(a, b, acc);
    }
    const int col = nt * 16 + (lane & 15);
    const int rb  = m0 + ((lane & 16) ? 8 : 0);
#pragma unroll
    for (int v = 0; v < 8; ++v) Out[(size_t)(rb + v) * C_DIM + col] = acc[v];
}

// Out = (1-beta)*H + beta*(H @ W)   (GCN2Conv combine fused into epilogue)
__global__ __launch_bounds__(256) void gemm_gcn2_kernel(const float* __restrict__ H,
                                                        const __bf16* __restrict__ Wb,
                                                        float* __restrict__ Out,
                                                        float beta, int M) {
    const int wave = (blockIdx.x * blockDim.x + threadIdx.x) >> 5;
    const int lane = threadIdx.x & 31;
    const int m0 = (wave / NTILE) * 16;
    const int nt = wave % NTILE;
    if (m0 >= M) return;
    v8f acc = {};
#pragma unroll
    for (int kt = 0; kt < KT128; ++kt) {
        v16bf a = load_a_f32(H, m0, kt * 32, lane);
        v16bf b = load_b_frag(Wb, nt, kt, lane);
        acc = wmma_bf16(a, b, acc);
    }
    const int col = nt * 16 + (lane & 15);
    const int rb  = m0 + ((lane & 16) ? 8 : 0);
#pragma unroll
    for (int v = 0; v < 8; ++v) {
        const size_t idx = (size_t)(rb + v) * C_DIM + col;
        Out[idx] = (1.0f - beta) * H[idx] + beta * acc[v];
    }
}

// Fused LSTM cell: gates = X@WihT (+ Hin@WhhT) + bias; pointwise in epilogue.
// WihT/WhhT pre-packed fragment-major with N = 4C (32 n-tiles).
__global__ __launch_bounds__(256) void lstm_kernel(const float* __restrict__ X,
                                                   const float* __restrict__ Hin,
                                                   const float* __restrict__ Cin,
                                                   const __bf16* __restrict__ WihT,
                                                   const __bf16* __restrict__ WhhT,
                                                   const float* __restrict__ bias,
                                                   float* __restrict__ Hout,
                                                   float* __restrict__ Cout, int M) {
    const int wave = (blockIdx.x * blockDim.x + threadIdx.x) >> 5;
    const int lane = threadIdx.x & 31;
    const int m0 = (wave / NTILE) * 16;
    const int nt = wave % NTILE;
    if (m0 >= M) return;
    v8f acc[4] = {};
#pragma unroll
    for (int kt = 0; kt < KT128; ++kt) {
        v16bf a = load_a_f32(X, m0, kt * 32, lane);
#pragma unroll
        for (int g = 0; g < 4; ++g) {
            v16bf b = load_b_frag(WihT, g * NTILE + nt, kt, lane);
            acc[g] = wmma_bf16(a, b, acc[g]);
        }
    }
    if (Hin) {
#pragma unroll
        for (int kt = 0; kt < KT128; ++kt) {
            v16bf a = load_a_f32(Hin, m0, kt * 32, lane);
#pragma unroll
            for (int g = 0; g < 4; ++g) {
                v16bf b = load_b_frag(WhhT, g * NTILE + nt, kt, lane);
                acc[g] = wmma_bf16(a, b, acc[g]);
            }
        }
    }
    const int col = nt * 16 + (lane & 15);
    const float bi = bias[col];
    const float bf = bias[C_DIM + col];
    const float bg = bias[2 * C_DIM + col];
    const float bo = bias[3 * C_DIM + col];
    const int rb = m0 + ((lane & 16) ? 8 : 0);
#pragma unroll
    for (int v = 0; v < 8; ++v) {
        const size_t idx = (size_t)(rb + v) * C_DIM + col;
        const float ig = sigmoidf_(acc[0][v] + bi);
        const float fg = sigmoidf_(acc[1][v] + bf);
        const float gg = tanhf(acc[2][v] + bg);
        const float og = sigmoidf_(acc[3][v] + bo);
        const float cold = Cin ? Cin[idx] : 0.0f;
        const float cnew = fg * cold + ig * gg;
        if (Cout) Cout[idx] = cnew;
        if (Hout) Hout[idx] = og * tanhf(cnew);
    }
}

// ---- Elementwise / graph kernels ------------------------------------------

__global__ void fill_kernel(float* p, int n, float v) {
    int i = blockIdx.x * blockDim.x + threadIdx.x;
    if (i < n) p[i] = v;
}

__global__ void count_deg_kernel(const int* __restrict__ dst, float* __restrict__ deg) {
    int e = blockIdx.x * blockDim.x + threadIdx.x;
    if (e < E_EDGES) atomAddF(&deg[dst[e]], 1.0f);
}

__global__ void dinv_kernel(const float* __restrict__ deg, float* __restrict__ dinv) {
    int i = blockIdx.x * blockDim.x + threadIdx.x;
    if (i < N_NODES) dinv[i] = rsqrtf(deg[i]);   // deg >= 1 (self loop)
}

// z = dinv[n]^2 * xw + bias[c]   (self-loop term + bias of GCNConv)
__global__ void zinit_kernel(const float* __restrict__ xw, const float* __restrict__ dinv,
                             const float* __restrict__ bias, float* __restrict__ z) {
    int i = blockIdx.x * blockDim.x + threadIdx.x;
    if (i < N_NODES * C_DIM) {
        int n = i >> 7, c = i & 127;
        float di = dinv[n];
        z[i] = di * di * xw[i] + bias[c];
    }
}

// z[dst] += dinv[s]*dinv[d] * xw[src]   (32 lanes x float4 per edge)
__global__ void gcn_scatter_kernel(const int* __restrict__ src, const int* __restrict__ dst,
                                   const float* __restrict__ dinv, const float* __restrict__ xw,
                                   float* __restrict__ z) {
    long gid = (long)blockIdx.x * blockDim.x + threadIdx.x;
    int e = (int)(gid >> 5);
    if (e >= E_EDGES) return;
    int q = ((int)gid & 31) << 2;
    int s = src[e], d = dst[e];
    float nrm = dinv[s] * dinv[d];
    const float4 v = *(const float4*)(xw + (size_t)s * C_DIM + q);
    float* zp = z + (size_t)d * C_DIM + q;
    atomAddF(zp + 0, nrm * v.x);
    atomAddF(zp + 1, nrm * v.y);
    atomAddF(zp + 2, nrm * v.z);
    atomAddF(zp + 3, nrm * v.w);
}

// agg[dst] += z[src]   (GCN2Conv aggregation)
__global__ void agg_scatter_kernel(const int* __restrict__ src, const int* __restrict__ dst,
                                   const float* __restrict__ z, float* __restrict__ agg) {
    long gid = (long)blockIdx.x * blockDim.x + threadIdx.x;
    int e = (int)(gid >> 5);
    if (e >= E_EDGES) return;
    int q = ((int)gid & 31) << 2;
    int s = src[e], d = dst[e];
    const float4 v = *(const float4*)(z + (size_t)s * C_DIM + q);
    float* ap = agg + (size_t)d * C_DIM + q;
    atomAddF(ap + 0, v.x);
    atomAddF(ap + 1, v.y);
    atomAddF(ap + 2, v.z);
    atomAddF(ap + 3, v.w);
}

// agg = (1-ALPHA)*agg + ALPHA*x0   (in place -> h buffer)
__global__ void combine_kernel(float* __restrict__ agg, const float* __restrict__ x0, int n) {
    int i = blockIdx.x * blockDim.x + threadIdx.x;
    if (i < n) agg[i] = 0.9f * agg[i] + 0.1f * x0[i];
}

// BatchNorm: per-channel partial sums, 256 rows per block
__global__ __launch_bounds__(128) void bn_partial_kernel(const float* __restrict__ z,
                                                         float* __restrict__ mu, float* __restrict__ m2) {
    int c = threadIdx.x;
    int r0 = blockIdx.x * 256;
    int r1 = r0 + 256; if (r1 > N_NODES) r1 = N_NODES;
    float s = 0.0f, s2 = 0.0f;
    for (int r = r0; r < r1; ++r) {
        float v = z[(size_t)r * C_DIM + c];
        s += v; s2 += v * v;
    }
    atomAddF(&mu[c], s);
    atomAddF(&m2[c], s2);
}

__global__ void bn_apply_kernel(const float* __restrict__ mu, const float* __restrict__ m2,
                                const float* __restrict__ gamma, const float* __restrict__ betap,
                                float* __restrict__ z) {
    int i = blockIdx.x * blockDim.x + threadIdx.x;
    if (i < N_NODES * C_DIM) {
        int c = i & 127;
        const float invN = 1.0f / (float)N_NODES;
        float mean = mu[c] * invN;
        float var  = m2[c] * invN - mean * mean;
        float inv  = rsqrtf(var + 1e-5f);
        float v = gamma[c] * (z[i] - mean) * inv + betap[c];
        z[i] = v > 0.0f ? v : 0.0f;
    }
}

__global__ void add_vec_kernel(const float* __restrict__ a, const float* __restrict__ b,
                               float* __restrict__ o, int n) {
    int i = blockIdx.x * blockDim.x + threadIdx.x;
    if (i < n) o[i] = a[i] + b[i];
}

// ---- Host orchestration ----------------------------------------------------

extern "C" void kernel_launch(void* const* d_in, const int* in_sizes, int n_in,
                              void* d_out, int out_size, void* d_ws, size_t ws_size,
                              hipStream_t stream) {
    (void)in_sizes; (void)n_in; (void)out_size; (void)ws_size;

    const float* x_seq    = (const float*)d_in[0];
    const int*   eidx     = (const int*)d_in[1];
    const float* lin0_w   = (const float*)d_in[2];
    const float* lin0_b   = (const float*)d_in[3];
    const float* conv_w1  = (const float*)d_in[4];
    const float* rec_Wih  = (const float*)d_in[5];
    // d_in[6] rec_Whh: unused (hidden state is zeroed each step in reference)
    const float* rec_bih  = (const float*)d_in[7];
    const float* rec_bhh  = (const float*)d_in[8];
    const float* feat_Wih = (const float*)d_in[9];
    const float* feat_Whh = (const float*)d_in[10];
    const float* feat_bih = (const float*)d_in[11];
    const float* feat_bhh = (const float*)d_in[12];
    const float* bn_gamma = (const float*)d_in[13];
    const float* bn_beta  = (const float*)d_in[14];
    float* out = (float*)d_out;

    const int NC = N_NODES * C_DIM;
    const int CC = C_DIM * C_DIM;

    char* w = (char*)d_ws;
    auto alloc = [&](size_t bytes) -> char* {
        char* p = w; w += (bytes + 255) & ~(size_t)255; return p;
    };
    float*  xw      = (float*)alloc((size_t)NC * 4);
    float*  x0      = (float*)alloc((size_t)NC * 4);
    float*  agg     = (float*)alloc((size_t)NC * 4);
    float*  deg     = (float*)alloc((size_t)N_NODES * 4);
    float*  dinv    = (float*)alloc((size_t)N_NODES * 4);
    float*  mu      = (float*)alloc(128 * 4);
    float*  m2      = (float*)alloc(128 * 4);
    float*  cells   = (float*)alloc((size_t)2 * CC * 4);
    float*  w1fA    = (float*)alloc((size_t)2 * CC * 4);
    float*  w1fB    = (float*)alloc((size_t)2 * CC * 4);
    __bf16* w1bf    = (__bf16*)alloc((size_t)2 * CC * 2);
    __bf16* lin0bf  = (__bf16*)alloc((size_t)CC * 2);
    __bf16* recWihT = (__bf16*)alloc((size_t)2 * C_DIM * FOURC * 2);
    __bf16* featWihT= (__bf16*)alloc((size_t)C_DIM * FOURC * 2);
    __bf16* featWhhT= (__bf16*)alloc((size_t)C_DIM * FOURC * 2);
    float*  recBias = (float*)alloc((size_t)2 * FOURC * 4);
    float*  featBias= (float*)alloc(FOURC * 4);
    // phase-2 buffers alias phase-1 scratch (disjoint lifetimes)
    float* hA = xw, * hB = x0, * cbuf = agg;

    const dim3 B256(256);
    auto blk = [](int n) { return dim3((n + 255) / 256); };
    const dim3 gemmBig(3125);   // 50000/16 * 8 tiles / 8 waves
    const dim3 gemmSm(8);       // 128/16 * 8 tiles / 8 waves
    const dim3 scat(E_EDGES * 32 / 256);

    // ---- one-time prep: pack all weights into fragment-major bf16 -------
    pack_b_kernel<<<blk(CC), B256, 0, stream>>>(lin0_w, lin0bf, C_DIM, C_DIM, 0);
    for (int l = 1; l <= 2; ++l) {
        pack_b_kernel<<<blk(FOURC * C_DIM), B256, 0, stream>>>(
            rec_Wih + (size_t)l * FOURC * C_DIM,
            recWihT + (size_t)(l - 1) * C_DIM * FOURC, C_DIM, FOURC, 1);
        add_vec_kernel<<<blk(FOURC), B256, 0, stream>>>(
            rec_bih + (size_t)l * FOURC, rec_bhh + (size_t)l * FOURC,
            recBias + (size_t)(l - 1) * FOURC, FOURC);
    }
    pack_b_kernel<<<blk(FOURC * C_DIM), B256, 0, stream>>>(feat_Wih, featWihT, C_DIM, FOURC, 1);
    pack_b_kernel<<<blk(FOURC * C_DIM), B256, 0, stream>>>(feat_Whh, featWhhT, C_DIM, FOURC, 1);
    add_vec_kernel<<<blk(FOURC), B256, 0, stream>>>(feat_bih, feat_bhh, featBias, FOURC);
    fill_kernel<<<blk(2 * CC), B256, 0, stream>>>(cells, 2 * CC, 0.0f);
    hipMemcpyAsync(w1fA, conv_w1, (size_t)2 * CC * 4, hipMemcpyDeviceToDevice, stream);

    const float beta_i[2] = { logf(1.5f), logf(1.25f) };   // log(THETA/(i+1)+1)
    float* w1cur = w1fA, * w1nxt = w1fB;

    // ---- timestep loop --------------------------------------------------
    for (int t = 0; t < 5; ++t) {
        float* zt = out + (size_t)t * NC;
        const int* srcp = eidx + ((size_t)t * 2 + 0) * E_EDGES;
        const int* dstp = eidx + ((size_t)t * 2 + 1) * E_EDGES;

        // evolve conv weights through their LSTMs (h input = 0)
        for (int i = 0; i < 2; ++i) {
            lstm_kernel<<<gemmSm, B256, 0, stream>>>(
                w1cur + (size_t)i * CC, nullptr, cells + (size_t)i * CC,
                recWihT + (size_t)i * C_DIM * FOURC, nullptr,
                recBias + (size_t)i * FOURC,
                w1nxt + (size_t)i * CC, cells + (size_t)i * CC, C_DIM);
        }
        // pack evolved conv weights (fragment-major bf16) for the GCN2 GEMMs
        pack_b_kernel<<<blk(CC), B256, 0, stream>>>(w1nxt, w1bf, C_DIM, C_DIM, 0);
        pack_b_kernel<<<blk(CC), B256, 0, stream>>>(w1nxt + CC, w1bf + CC, C_DIM, C_DIM, 0);
        { float* tmp = w1cur; w1cur = w1nxt; w1nxt = tmp; }

        // GCNConv: xw = x @ W, degree norm, self-loop init, edge scatter
        gemm_xw_kernel<<<gemmBig, B256, 0, stream>>>(x_seq + (size_t)t * NC, lin0bf, xw, N_NODES);
        fill_kernel<<<blk(N_NODES), B256, 0, stream>>>(deg, N_NODES, 1.0f);
        count_deg_kernel<<<blk(E_EDGES), B256, 0, stream>>>(dstp, deg);
        dinv_kernel<<<blk(N_NODES), B256, 0, stream>>>(deg, dinv);
        zinit_kernel<<<blk(NC), B256, 0, stream>>>(xw, dinv, lin0_b, zt);
        gcn_scatter_kernel<<<scat, B256, 0, stream>>>(srcp, dstp, dinv, xw, zt);
        hipMemcpyAsync(x0, zt, (size_t)NC * 4, hipMemcpyDeviceToDevice, stream);

        // two GCN2Conv layers with evolved weights
        for (int i = 0; i < 2; ++i) {
            fill_kernel<<<blk(NC), B256, 0, stream>>>(agg, NC, 0.0f);
            agg_scatter_kernel<<<scat, B256, 0, stream>>>(srcp, dstp, zt, agg);
            combine_kernel<<<blk(NC), B256, 0, stream>>>(agg, x0, NC);
            gemm_gcn2_kernel<<<gemmBig, B256, 0, stream>>>(agg, w1bf + (size_t)i * CC,
                                                           zt, beta_i[i], N_NODES);
            if (i == 0) {
                fill_kernel<<<blk(256), B256, 0, stream>>>(mu, 128, 0.0f);
                fill_kernel<<<blk(256), B256, 0, stream>>>(m2, 128, 0.0f);
                bn_partial_kernel<<<dim3(196), dim3(128), 0, stream>>>(zt, mu, m2);
                bn_apply_kernel<<<blk(NC), B256, 0, stream>>>(mu, m2, bn_gamma, bn_beta, zt);
            }
        }
    }

    // ---- future-prediction LSTM scan over z_seq[0..3] -------------------
    fill_kernel<<<blk(NC), B256, 0, stream>>>(cbuf, NC, 0.0f);
    float* hprev = nullptr;
    for (int s = 0; s < 4; ++s) {
        float* hout = (s & 1) ? hB : hA;
        lstm_kernel<<<gemmBig, B256, 0, stream>>>(
            out + (size_t)s * NC, hprev, cbuf,
            featWihT, featWhhT, featBias, hout, cbuf, N_NODES);
        hprev = hout;
    }
    // final step: x = ys[-1] = h, hidden input zeroed, c = final cell
    lstm_kernel<<<gemmBig, B256, 0, stream>>>(
        hprev, nullptr, cbuf, featWihT, featWhhT, featBias,
        out + (size_t)4 * NC, nullptr, N_NODES);
}
